// Attention_28913719837506
// MI455X (gfx1250) — matrix operand order repaired and checked by reading, not compiled, not run on hardware
//
#include <hip/hip_runtime.h>
#include <hip/hip_bf16.h>
#include <math.h>

typedef __bf16 bf16;
typedef __attribute__((ext_vector_type(16))) __bf16 v16bf;
typedef __attribute__((ext_vector_type(8)))  float  v8f;

union Frag16 { v16bf v; uint4 u[2]; };

__device__ __forceinline__ v8f wmma_bf16(const v16bf& a, const v16bf& b, const v8f& c) {
    // D = A(16x32 bf16) * B(32x16 bf16) + C(16x16 f32)
    return __builtin_amdgcn_wmma_f32_16x16x32_bf16(
        /*neg_a=*/false, a, /*neg_b=*/false, b,
        /*c_mod=*/(short)0, c, /*reuse_a=*/false, /*reuse_b=*/false);
}

// Async global->LDS copy, 16B per lane, tracked by ASYNCcnt.
// Low 32 bits of a flat LDS address are the LDS byte offset (ISA 10.2).
__device__ __forceinline__ void async_cp16(void* lds, const void* g) {
    asm volatile("global_load_async_to_lds_b128 %0, %1, off"
                 :: "v"((unsigned)(uintptr_t)lds), "v"(g)
                 : "memory");
}
#define WAIT_ASYNC(n) asm volatile("s_wait_asynccnt " #n ::: "memory")

#define HEADS    16
#define DHEAD    64
#define DIM      1024
#define NTOK     2048
#define BATCH    2
#define ROWS     (BATCH * NTOK)   // 4096

// ---------------------------------------------------------------------------
// K0: transpose fp32 weight [K=1024, N] -> bf16 [N, K] (K-major for WMMA B)
// ---------------------------------------------------------------------------
__global__ void k_transpose_bf16(const float* __restrict__ src, bf16* __restrict__ dst,
                                 int N, long total) {
    long idx = (long)blockIdx.x * 256 + threadIdx.x;   // idx = n*1024 + k
    if (idx >= total) return;
    int  k = (int)(idx & 1023);
    long n = idx >> 10;
    dst[idx] = (bf16)src[(long)k * N + n];
}

// ---------------------------------------------------------------------------
// K1: RMSNorm row -> bf16   xn = x / max(||x||,eps) * sqrt(dim) * gamma
// ---------------------------------------------------------------------------
__global__ __launch_bounds__(256) void k_rmsnorm(const float* __restrict__ x,
                                                 const float* __restrict__ gamma,
                                                 bf16* __restrict__ xn) {
    int row = blockIdx.x;
    const float4* x4 = (const float4*)(x + (size_t)row * DIM);
    const float4* g4 = (const float4*)gamma;
    float4 v = x4[threadIdx.x];
    float ss = v.x * v.x + v.y * v.y + v.z * v.z + v.w * v.w;
#pragma unroll
    for (int m = 1; m < 32; m <<= 1) ss += __shfl_xor(ss, m);
    __shared__ float red[8];
    if ((threadIdx.x & 31) == 0) red[threadIdx.x >> 5] = ss;
    __syncthreads();
    float tot = 0.f;
#pragma unroll
    for (int i = 0; i < 8; i++) tot += red[i];
    float scale = 32.0f / fmaxf(sqrtf(tot), 1e-12f);   // sqrt(1024)=32
    float4 g = g4[threadIdx.x];
    size_t o = (size_t)row * DIM + threadIdx.x * 4;
    xn[o + 0] = (bf16)(v.x * scale * g.x);
    xn[o + 1] = (bf16)(v.y * scale * g.y);
    xn[o + 2] = (bf16)(v.z * scale * g.z);
    xn[o + 3] = (bf16)(v.w * scale * g.w);
}

// ---------------------------------------------------------------------------
// K2: mix / gates  (sigmoid(xn @ W + b)), W is [1024,16]; 32 outputs per row
// ---------------------------------------------------------------------------
__global__ __launch_bounds__(256) void k_mixgates(const bf16* __restrict__ xn,
                                                  const float* __restrict__ w_mix,
                                                  const float* __restrict__ b_mix,
                                                  const float* __restrict__ w_gates,
                                                  const float* __restrict__ b_gates,
                                                  float* __restrict__ mix,
                                                  float* __restrict__ gates) {
    int row = blockIdx.x;
    int t = threadIdx.x;
    int o = t >> 3;            // 0..31 : 0..15 = mix heads, 16..31 = gate heads
    int part = t & 7;
    int head = o & 15;
    const float* W = (o < 16) ? w_mix : w_gates;
    const bf16* xr = xn + (size_t)row * DIM;
    float s = 0.f;
    int k0 = part * 128;
    for (int k = k0; k < k0 + 128; k++) s += (float)xr[k] * W[(size_t)k * HEADS + head];
    __shared__ float red[256];
    red[t] = s;
    __syncthreads();
    if (part == 0) {
        float tot = 0.f;
#pragma unroll
        for (int i = 0; i < 8; i++) tot += red[o * 8 + i];
        if (o < 16) {
            tot += b_mix[head];
            mix[(size_t)row * HEADS + head] = 1.f / (1.f + __expf(-tot));
        } else {
            tot += b_gates[head];
            gates[(size_t)row * HEADS + head] = 1.f / (1.f + __expf(-tot));
        }
    }
}

// ---------------------------------------------------------------------------
// K3: bf16 WMMA GEMM  C[M,N] = A[M,K(bf16)] * BT[N,K(bf16)]
//     128x128 block tile, double-buffered async LDS staging, k-step 32.
//     8 waves, each wave 32x64 (2x4 frags). Last iteration peeled.
//     mode 0: fp32 C -> outf ;  mode 1: QKV epilogue (q/k bf16, v fp32)
// ---------------------------------------------------------------------------
#define GSTRIDE 40   // LDS row stride in bf16 (64B data + 16B pad)

__global__ __launch_bounds__(256) void k_gemm(const bf16* __restrict__ A,
                                              const bf16* __restrict__ BT,
                                              int M, int N, int K, int mode,
                                              float* __restrict__ outf,
                                              bf16* __restrict__ qout,
                                              bf16* __restrict__ kout,
                                              float* __restrict__ vout) {
    __shared__ alignas(128) bf16 Asm[2][128 * GSTRIDE];
    __shared__ alignas(128) bf16 Bsm[2][128 * GSTRIDE];
    int t = threadIdx.x;
    int lane = t & 31, wave = t >> 5;
    int half = lane >> 4, l15 = lane & 15;
    int wm = wave & 3, wn = wave >> 2;
    int m0 = blockIdx.x * 128;
    int n0 = blockIdx.y * 128;

    v8f acc[2][4];
    v8f zero = {};
#pragma unroll
    for (int im = 0; im < 2; im++)
#pragma unroll
        for (int jn = 0; jn < 4; jn++) acc[im][jn] = zero;

    // stage one 128x32 A tile and B tile: 512 chunks of 16B each, 2/thread
    auto stage = [&](int buf, int k0) {
#pragma unroll
        for (int i = 0; i < 2; i++) {
            int c = i * 256 + t;
            int row = c >> 2, sub = c & 3;
            async_cp16(&Asm[buf][row * GSTRIDE + sub * 8],
                       A + (size_t)(m0 + row) * K + k0 + sub * 8);
            async_cp16(&Bsm[buf][row * GSTRIDE + sub * 8],
                       BT + (size_t)(n0 + row) * K + k0 + sub * 8);
        }
    };

    // one k-step: batch-load all fragments, then 8 WMMAs
    auto compute = [&](int cur) {
        Frag16 a[2];
#pragma unroll
        for (int im = 0; im < 2; im++) {
            const bf16* p = &Asm[cur][(wm * 32 + im * 16 + l15) * GSTRIDE + half * 8];
            a[im].u[0] = *(const uint4*)p;
            a[im].u[1] = *(const uint4*)(p + 16);
        }
        Frag16 bfr[4];
#pragma unroll
        for (int jn = 0; jn < 4; jn++) {
            const bf16* p = &Bsm[cur][(wn * 64 + jn * 16 + l15) * GSTRIDE + half * 8];
            bfr[jn].u[0] = *(const uint4*)p;
            bfr[jn].u[1] = *(const uint4*)(p + 16);
        }
#pragma unroll
        for (int jn = 0; jn < 4; jn++) {
            acc[0][jn] = wmma_bf16(a[0].v, bfr[jn].v, acc[0][jn]);
            acc[1][jn] = wmma_bf16(a[1].v, bfr[jn].v, acc[1][jn]);
        }
    };

    stage(0, 0);
    const int NS = K / 32;
    for (int ks = 0; ks < NS - 1; ks++) {          // steady state (last peeled)
        stage((ks + 1) & 1, (ks + 1) * 32);        // prefetch next (4 async/wave)
        WAIT_ASYNC(4);                             // current tile resident
        __syncthreads();
        compute(ks & 1);
        __syncthreads();                           // protect buffer re-stage
    }
    WAIT_ASYNC(0);
    __syncthreads();
    compute((NS - 1) & 1);

#pragma unroll
    for (int im = 0; im < 2; im++)
#pragma unroll
        for (int jn = 0; jn < 4; jn++)
#pragma unroll
            for (int r = 0; r < 8; r++) {
                int row = m0 + wm * 32 + im * 16 + r + 8 * half;
                int col = n0 + wn * 64 + jn * 16 + l15;
                float val = acc[im][jn][r];
                if (mode == 0) {
                    outf[(size_t)row * N + col] = val;
                } else {
                    int sec = col >> 10, rem = col & 1023;
                    int h = rem >> 6, d = rem & 63;
                    int bb = row >> 11, nn = row & 2047;
                    size_t idx = (((size_t)(bb * HEADS + h)) * NTOK + nn) * DHEAD + d;
                    if (sec == 0)      qout[idx] = (bf16)(val * 0.125f);  // fold 1/sqrt(64)
                    else if (sec == 1) kout[idx] = (bf16)val;
                    else               vout[idx] = val;                   // orig_v (fp32 out)
                }
            }
}

// ---------------------------------------------------------------------------
// K4: value-residual lerp -> transposed bf16 V  vT[b,h,d,n]
// ---------------------------------------------------------------------------
__global__ __launch_bounds__(256) void k_lerp(const float* __restrict__ vsrc,   // [b,h,n,d]
                                              const float* __restrict__ vres,   // [b,h,n,d]
                                              const float* __restrict__ mix,    // [b*n,16]
                                              bf16* __restrict__ vT) {          // [b,h,d,n]
    long idx = (long)blockIdx.x * 256 + threadIdx.x;   // idx = ((bh*64)+d)*2048 + nn
    int  nn = (int)(idx & 2047);
    long t  = idx >> 11;
    int  d  = (int)(t & 63);
    long bh = t >> 6;
    int  b  = (int)(bh >> 4), h = (int)(bh & 15);
    size_t si = ((size_t)bh * NTOK + nn) * DHEAD + d;
    float v = vsrc[si], r = vres[si];
    float m = mix[((size_t)b * NTOK + nn) * HEADS + h];
    vT[idx] = (bf16)(v + m * (r - v));
}

// ---------------------------------------------------------------------------
// K5: flash attention, double-buffered async K/V tiles in LDS, gated epilogue
//     grid (n/128, b*h), 8 waves x 16 query rows, 64-key tiles
// ---------------------------------------------------------------------------
#define ASTRIDE 72   // LDS row stride in bf16 (128B data + 16B pad)

__global__ __launch_bounds__(256) void k_attn(const bf16* __restrict__ Q,   // [b,h,n,d]
                                              const bf16* __restrict__ Km,  // [b,h,n,d]
                                              const bf16* __restrict__ VT,  // [b,h,d,n]
                                              const float* __restrict__ gates,
                                              bf16* __restrict__ out) {     // [b*n, 1024]
    __shared__ alignas(128) bf16 Ktile[2][64 * ASTRIDE];
    __shared__ alignas(128) bf16 Vtile[2][64 * ASTRIDE];
    __shared__ alignas(128) bf16 ptile[8 * 16 * ASTRIDE];
    int t = threadIdx.x;
    int lane = t & 31, wave = t >> 5;
    int half = lane >> 4, l15 = lane & 15;
    int bh = blockIdx.y, b = bh >> 4, h = bh & 15;
    const bf16* q  = Q  + (size_t)bh * NTOK * DHEAD;
    const bf16* k  = Km + (size_t)bh * NTOK * DHEAD;
    const bf16* vT = VT + (size_t)bh * DHEAD * NTOK;
    int q0 = blockIdx.x * 128 + wave * 16;

    // Q A-fragments resident for whole kernel (k-dims 0..31 and 32..63)
    Frag16 qa, qb;
    {
        const bf16* p = q + (size_t)(q0 + l15) * DHEAD + half * 8;
        qa.u[0] = *(const uint4*)p;  qa.u[1] = *(const uint4*)(p + 16);
        p += 32;
        qb.u[0] = *(const uint4*)p;  qb.u[1] = *(const uint4*)(p + 16);
    }

    v8f o[4];
    v8f zero = {};
#pragma unroll
    for (int jn = 0; jn < 4; jn++) o[jn] = zero;
    float mi[8], li[8];
#pragma unroll
    for (int r = 0; r < 8; r++) { mi[r] = -1e30f; li[r] = 0.f; }

    bf16* pbase = ptile + wave * 16 * ASTRIDE;

    // stage 64x64 K tile (8KB) + 64x64 V tile (8KB): 512 chunks each, 2/thread
    auto stage = [&](int buf, int kt) {
#pragma unroll
        for (int i = 0; i < 2; i++) {
            int c = i * 256 + t;
            int row = c >> 3, sub = c & 7;
            async_cp16(&Ktile[buf][row * ASTRIDE + sub * 8],
                       k + (size_t)(kt + row) * DHEAD + sub * 8);
            async_cp16(&Vtile[buf][row * ASTRIDE + sub * 8],
                       vT + (size_t)row * NTOK + kt + sub * 8);
        }
    };

    // one 64-key tile step against buffer `cur`
    auto tile_step = [&](int cur) {
        // ---- S = Q * K^T : batch-load K frags per k-half, then WMMAs
        v8f s[4];
        Frag16 kf[4];
#pragma unroll
        for (int j = 0; j < 4; j++) {
            const bf16* p = &Ktile[cur][(j * 16 + l15) * ASTRIDE + half * 8];
            kf[j].u[0] = *(const uint4*)p;
            kf[j].u[1] = *(const uint4*)(p + 16);
        }
#pragma unroll
        for (int j = 0; j < 4; j++) s[j] = wmma_bf16(qa.v, kf[j].v, zero);
#pragma unroll
        for (int j = 0; j < 4; j++) {
            const bf16* p = &Ktile[cur][(j * 16 + l15) * ASTRIDE + 32 + half * 8];
            kf[j].u[0] = *(const uint4*)p;
            kf[j].u[1] = *(const uint4*)(p + 16);
        }
#pragma unroll
        for (int j = 0; j < 4; j++) s[j] = wmma_bf16(qb.v, kf[j].v, s[j]);

        // ---- online softmax: row max over 64 cols (4 frags x 16 lanes)
        float alpha[8];
#pragma unroll
        for (int r = 0; r < 8; r++) {
            float mx = fmaxf(fmaxf(s[0][r], s[1][r]), fmaxf(s[2][r], s[3][r]));
#pragma unroll
            for (int msk = 1; msk < 16; msk <<= 1) mx = fmaxf(mx, __shfl_xor(mx, msk));
            float mn = fmaxf(mi[r], mx);
            alpha[r] = __expf(mi[r] - mn);
            mi[r] = mn;
            li[r] *= alpha[r];
        }
#pragma unroll
        for (int jn = 0; jn < 4; jn++)
#pragma unroll
            for (int r = 0; r < 8; r++) o[jn][r] *= alpha[r];

        float rs[8];
#pragma unroll
        for (int r = 0; r < 8; r++) rs[r] = 0.f;
#pragma unroll
        for (int j = 0; j < 4; j++)
#pragma unroll
            for (int r = 0; r < 8; r++) {
                float pr = __expf(s[j][r] - mi[r]);
                s[j][r] = pr;
                rs[r] += pr;
            }
#pragma unroll
        for (int r = 0; r < 8; r++) {
#pragma unroll
            for (int msk = 1; msk < 16; msk <<= 1) rs[r] += __shfl_xor(rs[r], msk);
            li[r] += rs[r];
        }

        // ---- stage P (C-layout) through per-wave LDS, reload as A-fragments
#pragma unroll
        for (int j = 0; j < 4; j++)
#pragma unroll
            for (int r = 0; r < 8; r++)
                pbase[(r + 8 * half) * ASTRIDE + j * 16 + l15] = (bf16)s[j][r];
        asm volatile("s_wait_dscnt 0" ::: "memory");
        Frag16 pa, pb;
        {
            const bf16* p = pbase + l15 * ASTRIDE + half * 8;
            pa.u[0] = *(const uint4*)p;  pa.u[1] = *(const uint4*)(p + 16);
            p += 32;
            pb.u[0] = *(const uint4*)p;  pb.u[1] = *(const uint4*)(p + 16);
        }

        // ---- O += P * V : batch-load V frags per key-half, then WMMAs
        Frag16 vf[4];
#pragma unroll
        for (int jn = 0; jn < 4; jn++) {
            const bf16* p = &Vtile[cur][(jn * 16 + l15) * ASTRIDE + half * 8];
            vf[jn].u[0] = *(const uint4*)p;
            vf[jn].u[1] = *(const uint4*)(p + 16);
        }
#pragma unroll
        for (int jn = 0; jn < 4; jn++) o[jn] = wmma_bf16(pa.v, vf[jn].v, o[jn]);
#pragma unroll
        for (int jn = 0; jn < 4; jn++) {
            const bf16* p = &Vtile[cur][(jn * 16 + l15) * ASTRIDE + 32 + half * 8];
            vf[jn].u[0] = *(const uint4*)p;
            vf[jn].u[1] = *(const uint4*)(p + 16);
        }
#pragma unroll
        for (int jn = 0; jn < 4; jn++) o[jn] = wmma_bf16(pb.v, vf[jn].v, o[jn]);
    };

    stage(0, 0);
    const int NS = NTOK / 64;
    for (int ks = 0; ks < NS - 1; ks++) {          // steady state (last peeled)
        stage((ks + 1) & 1, (ks + 1) * 64);        // prefetch next K/V (4 async/wave)
        WAIT_ASYNC(4);
        __syncthreads();
        tile_step(ks & 1);
        __syncthreads();                           // protect buffers before re-stage
    }
    WAIT_ASYNC(0);
    __syncthreads();
    tile_step((NS - 1) & 1);

    // ---- epilogue: gate, normalize by li, emit bf16 [b*n, h*64+d]
#pragma unroll
    for (int r = 0; r < 8; r++) {
        int tok = q0 + r + 8 * half;
        float g = gates[((size_t)b * NTOK + tok) * HEADS + h];
        float inv = g / li[r];
#pragma unroll
        for (int jn = 0; jn < 4; jn++)
            out[((size_t)b * NTOK + tok) * DIM + h * DHEAD + jn * 16 + l15] =
                (bf16)(o[jn][r] * inv);
    }
}

// ---------------------------------------------------------------------------
extern "C" void kernel_launch(void* const* d_in, const int* in_sizes, int n_in,
                              void* d_out, int out_size, void* d_ws, size_t ws_size,
                              hipStream_t stream) {
    const float* x       = (const float*)d_in[0];
    const float* vres    = (const float*)d_in[1];
    const float* gamma   = (const float*)d_in[2];
    const float* w_qkv   = (const float*)d_in[3];
    const float* w_mix   = (const float*)d_in[4];
    const float* b_mix   = (const float*)d_in[5];
    const float* w_gates = (const float*)d_in[6];
    const float* b_gates = (const float*)d_in[7];
    const float* w_out   = (const float*)d_in[8];

    float* outp  = (float*)d_out;                          // [2,2048,1024]
    float* origv = outp + (size_t)BATCH * NTOK * DIM;      // [2,16,2048,64]

    char* ws = (char*)d_ws;
    bf16* wTqkv  = (bf16*)ws;  ws += (size_t)3 * DIM * DIM * 2;       // [3072,1024]
    bf16* wTout  = (bf16*)ws;  ws += (size_t)DIM * DIM * 2;           // [1024,1024]
    bf16* xn     = (bf16*)ws;  ws += (size_t)ROWS * DIM * 2;          // [4096,1024]
    float* mixb  = (float*)ws; ws += (size_t)ROWS * HEADS * 4;
    float* gateb = (float*)ws; ws += (size_t)ROWS * HEADS * 4;
    bf16* qbf    = (bf16*)ws;  ws += (size_t)ROWS * DIM * 2;          // [b,h,n,d]
    bf16* kbf    = (bf16*)ws;  ws += (size_t)ROWS * DIM * 2;
    bf16* vTb    = (bf16*)ws;  ws += (size_t)ROWS * DIM * 2;          // [b,h,d,n]
    bf16* attnbf = (bf16*)ws;  ws += (size_t)ROWS * DIM * 2;          // [4096,1024]

    // weights -> bf16, K-major
    {
        long tot = (long)3 * DIM * DIM;
        k_transpose_bf16<<<(int)((tot + 255) / 256), 256, 0, stream>>>(w_qkv, wTqkv, 3 * DIM, tot);
        tot = (long)DIM * DIM;
        k_transpose_bf16<<<(int)((tot + 255) / 256), 256, 0, stream>>>(w_out, wTout, DIM, tot);
    }
    k_rmsnorm<<<ROWS, 256, 0, stream>>>(x, gamma, xn);
    k_mixgates<<<ROWS, 256, 0, stream>>>(xn, w_mix, b_mix, w_gates, b_gates, mixb, gateb);
    // QKV projection (q scaled, k bf16, orig_v fp32 -> d_out)
    k_gemm<<<dim3(ROWS / 128, (3 * DIM) / 128), 256, 0, stream>>>(
        xn, wTqkv, ROWS, 3 * DIM, DIM, 1, nullptr, qbf, kbf, origv);
    // v lerp -> transposed bf16
    {
        long tot = (long)ROWS * DIM;
        k_lerp<<<(int)(tot / 256), 256, 0, stream>>>(origv, vres, mixb, vTb);
    }
    // flash attention + gating
    k_attn<<<dim3(NTOK / 128, BATCH * HEADS), 256, 0, stream>>>(qbf, kbf, vTb, gateb, attnbf);
    // output projection fp32 -> d_out
    k_gemm<<<dim3(ROWS / 128, DIM / 128), 256, 0, stream>>>(
        attnbf, wTout, ROWS, DIM, DIM, 0, outp, nullptr, nullptr, nullptr);
}
